// MultiHeadSelfAttentionBlock_40647570489684
// MI455X (gfx1250) — matrix operand order repaired and checked by reading, not compiled, not run on hardware
//
#include <hip/hip_runtime.h>

// ---------------------------------------------------------------------------
// MHA block for MI455X (gfx1250, wave32).  bf16 WMMA (16x16x32) w/ f32 accum.
// B=2, S=2048, D=1024, H=16, HD=64.
// ---------------------------------------------------------------------------

typedef __attribute__((ext_vector_type(16))) __bf16 v16bf;
typedef __attribute__((ext_vector_type(8)))  __bf16 v8bf;
typedef __attribute__((ext_vector_type(4)))  __bf16 v4bf;
typedef __attribute__((ext_vector_type(2)))  __bf16 v2bf;
typedef __attribute__((ext_vector_type(8)))  float  v8f;

#define WMMA_BF16(a, b, c) \
  __builtin_amdgcn_wmma_f32_16x16x32_bf16(false, (a), false, (b), (short)0, (c), false, false)

static __device__ __forceinline__ v16bf join8(v8bf lo, v8bf hi) {
  return __builtin_shufflevector(lo, hi, 0,1,2,3,4,5,6,7,8,9,10,11,12,13,14,15);
}
static __device__ __forceinline__ v16bf lds_load16(const __bf16* p) {
  return join8(*(const v8bf*)p, *(const v8bf*)(p + 8));   // 2x ds_load_b128
}
static __device__ __forceinline__ v4bf cvt4(float4 f) {
  v4bf p; p[0] = (__bf16)f.x; p[1] = (__bf16)f.y; p[2] = (__bf16)f.z; p[3] = (__bf16)f.w;
  return p;
}
static __device__ __forceinline__ float f4get(float4 f, int j) {
  float a[4] = {f.x, f.y, f.z, f.w};
  return a[j];   // j is a constant after unrolling
}

// ======================= Kernel 1: QKV projection ==========================
// C[4096,3072] = x[4096,1024] @ w_qkv[1024,3072] + b_qkv, scattered into
// q/k/v workspace in (b, h, s, hd) layout.  Block tile 128x64, 4 waves,
// each wave: 32 rows x 64 cols = 8 WMMA accumulators.
__global__ __launch_bounds__(128) void qkv_gemm_kernel(
    const float* __restrict__ x, const float* __restrict__ w,
    const float* __restrict__ bias,
    float* __restrict__ qbuf, float* __restrict__ kbuf, float* __restrict__ vbuf)
{
  __shared__ alignas(16) __bf16 As[128][32];   // [row][k]
  __shared__ alignas(16) __bf16 Bs[64][32];    // transposed: [col][k]
  const int tid  = threadIdx.x;
  const int wave = tid >> 5;
  const int lane = tid & 31;
  const int lm   = lane & 15;
  const int hi   = lane >> 4;
  const int rowbase = blockIdx.y * 128;
  const int colbase = blockIdx.x * 64;

  v8f acc[2][4];
#pragma unroll
  for (int st = 0; st < 2; ++st)
#pragma unroll
    for (int g = 0; g < 4; ++g)
#pragma unroll
      for (int i = 0; i < 8; ++i) acc[st][g][i] = 0.0f;

  // per-thread staging assignment (float4 granularity)
  const int ar = tid >> 3;                  // A row (step 16 per chunk)
  const int ac = (tid & 7) * 4;             // A k-col
  const int bg = (tid & 15) * 4;            // B col base
  const int bk = (tid >> 4) * 2;            // B k-pair base (step 16 per chunk)
  const float* aptr = x + (size_t)(rowbase + ar) * 1024 + ac;
  const float* bptr = w + (size_t)bk * 3072 + colbase + bg;

  for (int k0 = 0; k0 < 1024; k0 += 32) {
    float4 ta[8];
#pragma unroll
    for (int i = 0; i < 8; ++i)
      ta[i] = *(const float4*)(aptr + (size_t)(16 * i) * 1024);
    float4 tb0[2], tb1[2];
#pragma unroll
    for (int i = 0; i < 2; ++i) {
      tb0[i] = *(const float4*)(bptr + (size_t)(16 * i) * 3072);
      tb1[i] = *(const float4*)(bptr + (size_t)(16 * i + 1) * 3072);
    }
    if (k0 + 32 < 1024) {
      __builtin_prefetch(aptr + 32, 0, 3);
      __builtin_prefetch(bptr + (size_t)32 * 3072, 0, 3);
    }
#pragma unroll
    for (int i = 0; i < 8; ++i)
      *(v4bf*)&As[ar + 16 * i][ac] = cvt4(ta[i]);        // ds_store_b64
#pragma unroll
    for (int i = 0; i < 2; ++i) {
      int r = bk + 16 * i;
#pragma unroll
      for (int j = 0; j < 4; ++j) {                      // transpose via packed b32
        v2bf p; p[0] = (__bf16)f4get(tb0[i], j); p[1] = (__bf16)f4get(tb1[i], j);
        *(v2bf*)&Bs[bg + j][r] = p;
      }
    }
    __syncthreads();

    v16bf af[2];
#pragma unroll
    for (int st = 0; st < 2; ++st) {
      const __bf16* ap = &As[wave * 32 + st * 16 + lm][0];
      af[st] = join8(*(const v8bf*)(ap + 16 * hi), *(const v8bf*)(ap + 16 * hi + 8));
    }
#pragma unroll
    for (int g = 0; g < 4; ++g) {
      v16bf bf = lds_load16(&Bs[g * 16 + lm][16 * hi]);
      acc[0][g] = WMMA_BF16(af[0], bf, acc[0][g]);
      acc[1][g] = WMMA_BF16(af[1], bf, acc[1][g]);
    }
    __syncthreads();
    aptr += 32;
    bptr += (size_t)32 * 3072;
  }

#pragma unroll
  for (int st = 0; st < 2; ++st)
#pragma unroll
    for (int g = 0; g < 4; ++g)
#pragma unroll
      for (int i = 0; i < 8; ++i) {
        int r = rowbase + wave * 32 + st * 16 + i + 8 * hi;   // b*2048 + s
        int c = colbase + g * 16 + lm;                        // [0,3072)
        float val = acc[st][g][i] + bias[c];
        int b = r >> 11, s = r & 2047;
        int h = c / 192, within = c % 192;
        int which = within >> 6, hd = within & 63;
        float* dst = (which == 0) ? qbuf : ((which == 1) ? kbuf : vbuf);
        dst[(((size_t)(b * 16 + h)) * 2048 + s) * 64 + hd] = val;
      }
}

// ======================= Kernel 2: flash attention =========================
// Block: 64 query rows (4 waves x 16).  Block-uniform loop over 32-key tiles
// up to the block's causal limit; K and V^T staged cooperatively in LDS.
__global__ __launch_bounds__(128) void flash_attn_kernel(
    const float* __restrict__ qbuf, const float* __restrict__ kbuf,
    const float* __restrict__ vbuf, float* __restrict__ obuf)
{
  __shared__ alignas(16) __bf16 Ks[32][64];        // [key][kdim]
  __shared__ alignas(16) __bf16 Vt[64][32];        // transposed: [hd][key]
  __shared__ alignas(16) __bf16 Plds[4][16][32];   // per-wave P tile [row][key]
  const int tid  = threadIdx.x;
  const int wave = tid >> 5;
  const int lane = tid & 31;
  const int lm   = lane & 15;
  const int hi   = lane >> 4;
  const int bh   = blockIdx.x;                     // b*16 + h
  const int h    = bh & 15;
  const int qbB  = blockIdx.y * 64;
  const int qb   = qbB + wave * 16;

  const float* Q = qbuf + (size_t)bh * 2048 * 64;
  const float* K = kbuf + (size_t)bh * 2048 * 64;
  const float* V = vbuf + (size_t)bh * 2048 * 64;

  // Q fragments (A-layout, 16x32 each): kdim 0..31 and 32..63
  const float* Qrow = Q + (size_t)(qb + lm) * 64;
  v16bf qa0, qa1;
#pragma unroll
  for (int j = 0; j < 16; ++j) {
    int kk = j + 16 * hi;
    qa0[j] = (__bf16)Qrow[kk];
    qa1[j] = (__bf16)Qrow[32 + kk];
  }

  v8f o[4];
  float m8[8], l8[8];
#pragma unroll
  for (int g = 0; g < 4; ++g)
#pragma unroll
    for (int i = 0; i < 8; ++i) o[g][i] = 0.0f;
#pragma unroll
  for (int i = 0; i < 8; ++i) { m8[i] = -1e30f; l8[i] = 0.0f; }

  // staging assignment
  const int kkey = tid >> 4;            // K: key row (step 8 per chunk)
  const int kc   = (tid & 15) * 4;      // K: kdim col
  const int vkp  = (tid >> 4) * 2;      // V: key-pair base (step 16 per chunk)
  const int vc   = (tid & 15) * 4;      // V: hd col base

  const int kend = qbB + 64;            // block-uniform causal bound
  for (int kt = 0; kt < kend; kt += 32) {
    // ---- cooperative staging: K (row-major) and V (transposed) ----
    float4 tk[4];
#pragma unroll
    for (int i = 0; i < 4; ++i)
      tk[i] = *(const float4*)(K + (size_t)(kt + kkey + 8 * i) * 64 + kc);
    float4 tv0[2], tv1[2];
#pragma unroll
    for (int i = 0; i < 2; ++i) {
      int key = kt + vkp + 16 * i;
      tv0[i] = *(const float4*)(V + (size_t)key * 64 + vc);
      tv1[i] = *(const float4*)(V + (size_t)(key + 1) * 64 + vc);
    }
    if (kt + 32 < kend) {
      __builtin_prefetch(K + (size_t)(kt + 32 + kkey) * 64 + kc, 0, 3);
      __builtin_prefetch(V + (size_t)(kt + 32 + vkp) * 64 + vc, 0, 3);
    }
#pragma unroll
    for (int i = 0; i < 4; ++i)
      *(v4bf*)&Ks[kkey + 8 * i][kc] = cvt4(tk[i]);
#pragma unroll
    for (int i = 0; i < 2; ++i) {
      int r = vkp + 16 * i;
#pragma unroll
      for (int j = 0; j < 4; ++j) {
        v2bf p; p[0] = (__bf16)f4get(tv0[i], j); p[1] = (__bf16)f4get(tv1[i], j);
        *(v2bf*)&Vt[vc + j][r] = p;
      }
    }
    __syncthreads();

    // ---- S = Q @ K^T (two 16-key groups, HD=64 -> 2 WMMA each) ----
    v8f s0, s1;
#pragma unroll
    for (int i = 0; i < 8; ++i) { s0[i] = 0.0f; s1[i] = 0.0f; }
    s0 = WMMA_BF16(qa0, lds_load16(&Ks[lm][16 * hi]), s0);
    s0 = WMMA_BF16(qa1, lds_load16(&Ks[lm][32 + 16 * hi]), s0);
    s1 = WMMA_BF16(qa0, lds_load16(&Ks[16 + lm][16 * hi]), s1);
    s1 = WMMA_BF16(qa1, lds_load16(&Ks[16 + lm][32 + 16 * hi]), s1);

    // ---- scale + causal mask + online softmax ----
#pragma unroll
    for (int i = 0; i < 8; ++i) {
      int row  = qb + i + 8 * hi;
      int key0 = kt + lm;
      int key1 = key0 + 16;
      float x0 = (key0 > row) ? -1e9f : s0[i] * 0.125f;
      float x1 = (key1 > row) ? -1e9f : s1[i] * 0.125f;
      float t = fmaxf(x0, x1);
      t = fmaxf(t, __shfl_xor(t, 1, 32));
      t = fmaxf(t, __shfl_xor(t, 2, 32));
      t = fmaxf(t, __shfl_xor(t, 4, 32));
      t = fmaxf(t, __shfl_xor(t, 8, 32));
      float mnew  = fmaxf(m8[i], t);
      float alpha = __expf(m8[i] - mnew);
      float p0 = __expf(x0 - mnew);
      float p1 = __expf(x1 - mnew);
      float rs = p0 + p1;
      rs += __shfl_xor(rs, 1, 32);
      rs += __shfl_xor(rs, 2, 32);
      rs += __shfl_xor(rs, 4, 32);
      rs += __shfl_xor(rs, 8, 32);
      l8[i] = l8[i] * alpha + rs;
      m8[i] = mnew;
#pragma unroll
      for (int g = 0; g < 4; ++g) o[g][i] *= alpha;
      Plds[wave][i + 8 * hi][lm]      = (__bf16)p0;   // C-layout -> [row][key]
      Plds[wave][i + 8 * hi][lm + 16] = (__bf16)p1;
    }

    // ---- regather P as A fragment; O += P @ V ----
    v16bf pfrag = join8(*(const v8bf*)&Plds[wave][lm][16 * hi],
                        *(const v8bf*)&Plds[wave][lm][16 * hi + 8]);
#pragma unroll
    for (int g = 0; g < 4; ++g) {
      v16bf vf = lds_load16(&Vt[g * 16 + lm][16 * hi]);
      o[g] = WMMA_BF16(pfrag, vf, o[g]);
    }
    __syncthreads();
  }

  // ---- normalize + store to (b, s, d) workspace ----
  const int b = bh >> 4;
#pragma unroll
  for (int i = 0; i < 8; ++i) {
    float inv = 1.0f / l8[i];
    int row = qb + i + 8 * hi;
    size_t base = ((size_t)b * 2048 + row) * 1024 + h * 64 + lm;
#pragma unroll
    for (int g = 0; g < 4; ++g)
      obuf[base + g * 16] = o[g][i] * inv;
  }
}

// ======================= Kernel 3: output projection =======================
// out[4096,1024] = attn[4096,1024] @ w_out[1024,1024] + b_out
__global__ __launch_bounds__(128) void out_gemm_kernel(
    const float* __restrict__ a, const float* __restrict__ w,
    const float* __restrict__ bias, float* __restrict__ out)
{
  __shared__ alignas(16) __bf16 As[128][32];
  __shared__ alignas(16) __bf16 Bs[64][32];
  const int tid  = threadIdx.x;
  const int wave = tid >> 5;
  const int lane = tid & 31;
  const int lm   = lane & 15;
  const int hi   = lane >> 4;
  const int rowbase = blockIdx.y * 128;
  const int colbase = blockIdx.x * 64;

  v8f acc[2][4];
#pragma unroll
  for (int st = 0; st < 2; ++st)
#pragma unroll
    for (int g = 0; g < 4; ++g)
#pragma unroll
      for (int i = 0; i < 8; ++i) acc[st][g][i] = 0.0f;

  const int ar = tid >> 3;
  const int ac = (tid & 7) * 4;
  const int bg = (tid & 15) * 4;
  const int bk = (tid >> 4) * 2;
  const float* aptr = a + (size_t)(rowbase + ar) * 1024 + ac;
  const float* bptr = w + (size_t)bk * 1024 + colbase + bg;

  for (int k0 = 0; k0 < 1024; k0 += 32) {
    float4 ta[8];
#pragma unroll
    for (int i = 0; i < 8; ++i)
      ta[i] = *(const float4*)(aptr + (size_t)(16 * i) * 1024);
    float4 tb0[2], tb1[2];
#pragma unroll
    for (int i = 0; i < 2; ++i) {
      tb0[i] = *(const float4*)(bptr + (size_t)(16 * i) * 1024);
      tb1[i] = *(const float4*)(bptr + (size_t)(16 * i + 1) * 1024);
    }
    if (k0 + 32 < 1024) {
      __builtin_prefetch(aptr + 32, 0, 3);
      __builtin_prefetch(bptr + (size_t)32 * 1024, 0, 3);
    }
#pragma unroll
    for (int i = 0; i < 8; ++i)
      *(v4bf*)&As[ar + 16 * i][ac] = cvt4(ta[i]);
#pragma unroll
    for (int i = 0; i < 2; ++i) {
      int r = bk + 16 * i;
#pragma unroll
      for (int j = 0; j < 4; ++j) {
        v2bf p; p[0] = (__bf16)f4get(tb0[i], j); p[1] = (__bf16)f4get(tb1[i], j);
        *(v2bf*)&Bs[bg + j][r] = p;
      }
    }
    __syncthreads();

    v16bf af[2];
#pragma unroll
    for (int st = 0; st < 2; ++st) {
      const __bf16* ap = &As[wave * 32 + st * 16 + lm][0];
      af[st] = join8(*(const v8bf*)(ap + 16 * hi), *(const v8bf*)(ap + 16 * hi + 8));
    }
#pragma unroll
    for (int g = 0; g < 4; ++g) {
      v16bf bf = lds_load16(&Bs[g * 16 + lm][16 * hi]);
      acc[0][g] = WMMA_BF16(af[0], bf, acc[0][g]);
      acc[1][g] = WMMA_BF16(af[1], bf, acc[1][g]);
    }
    __syncthreads();
    aptr += 32;
    bptr += (size_t)32 * 1024;
  }

#pragma unroll
  for (int st = 0; st < 2; ++st)
#pragma unroll
    for (int g = 0; g < 4; ++g)
#pragma unroll
      for (int i = 0; i < 8; ++i) {
        int r = rowbase + wave * 32 + st * 16 + i + 8 * hi;
        int c = colbase + g * 16 + lm;
        out[(size_t)r * 1024 + c] = acc[st][g][i] + bias[c];
      }
}

// ============================== Launcher ===================================
extern "C" void kernel_launch(void* const* d_in, const int* in_sizes, int n_in,
                              void* d_out, int out_size, void* d_ws, size_t ws_size,
                              hipStream_t stream) {
  (void)in_sizes; (void)n_in; (void)out_size; (void)ws_size;
  const float* x     = (const float*)d_in[0];
  const float* mask  = (const float*)d_in[1];  // deterministic causal mask: computed analytically
  (void)mask;
  const float* w_qkv = (const float*)d_in[2];
  const float* b_qkv = (const float*)d_in[3];
  const float* w_out = (const float*)d_in[4];
  const float* b_out = (const float*)d_in[5];
  float* out = (float*)d_out;

  const size_t per = (size_t)2 * 16 * 2048 * 64;   // 4,194,304 floats each
  float* qb = (float*)d_ws;
  float* kb = qb + per;
  float* vb = kb + per;
  float* ab = vb + per;

  qkv_gemm_kernel<<<dim3(48, 32), 128, 0, stream>>>(x, w_qkv, b_qkv, qb, kb, vb);
  flash_attn_kernel<<<dim3(32, 32), 128, 0, stream>>>(qb, kb, vb, ab);
  out_gemm_kernel<<<dim3(16, 32), 128, 0, stream>>>(ab, w_out, b_out, out);
}
